// AttentionModule_42649025249282
// MI455X (gfx1250) — matrix operand order repaired and checked by reading, not compiled, not run on hardware
//
#include <hip/hip_runtime.h>

typedef __attribute__((ext_vector_type(16))) _Float16 v16h;
typedef __attribute__((ext_vector_type(8)))  float    v8f;
typedef __attribute__((ext_vector_type(8)))  unsigned int v8u;

#define BATCH 4
#define SEQ   8192
#define DIN   256
#define DH    128
#define ROWS  (BATCH * SEQ)

static __device__ __forceinline__ v8f wmma_f16(v16h a, v16h b, v8f c) {
  // D = A(16x32 f16) * B(32x16 f16) + C(16x16 f32)
  return __builtin_amdgcn_wmma_f32_16x16x32_f16(
      /*neg_a=*/false, a, /*neg_b=*/false, b,
      /*c_mod=*/(short)0, c, /*reuse_a=*/false, /*reuse_b=*/false);
}

// ---------------------------------------------------------------------------
// Kernel 1: QKV projection  out[r, n] = sum_k x[r,k] * W[k,n]   (f16 out)
// Block: 256 threads (8 waves). Tile: 128 rows x 128 cols, K chunked by 64.
// ---------------------------------------------------------------------------
__global__ __launch_bounds__(256) void qkv_proj_kernel(
    const float* __restrict__ x, const float* __restrict__ W,
    _Float16* __restrict__ out, float scale) {
  __shared__ __align__(32) _Float16 xs[128][64];   // x chunk, row-major
  __shared__ __align__(32) _Float16 Wt[128][64];   // W chunk, transposed [n][k]

  const int tid  = threadIdx.x;
  const int lane = tid & 31;
  const int wave = tid >> 5;
  const int m    = lane & 15;     // A-matrix row / B-matrix col within tile
  const int hi   = lane >> 4;     // lane half
  const int rowBase = blockIdx.x * 128;

  v8f acc[8] = {};

  for (int kb = 0; kb < DIN; kb += 64) {
    __syncthreads();  // protect previous chunk's LDS reads
    for (int i = tid; i < 128 * 64; i += 256) {
      int r = i >> 6, c = i & 63;
      xs[r][c] = (_Float16)x[(size_t)(rowBase + r) * DIN + kb + c];
    }
    for (int i = tid; i < 64 * 128; i += 256) {
      int k = i >> 7, n = i & 127;
      Wt[n][k] = (_Float16)W[(size_t)(kb + k) * DH + n];
    }
    __syncthreads();

    const int arow = wave * 16 + m;
#pragma unroll
    for (int kk = 0; kk < 2; ++kk) {
      // A fragment (16x32 f16): pairs of consecutive K per VGPR
      v8u au;
#pragma unroll
      for (int p = 0; p < 8; ++p) {
        int k = kk * 32 + hi * 16 + p * 2;
        au[p] = *(const unsigned int*)&xs[arow][k];
      }
      v16h a = __builtin_bit_cast(v16h, au);
#pragma unroll
      for (int nt = 0; nt < 8; ++nt) {
        // B fragment (32x16): 16 contiguous K halves from transposed W
        v16h b = *(const v16h*)&Wt[nt * 16 + m][kk * 32 + hi * 16];
        acc[nt] = wmma_f16(a, b, acc[nt]);
      }
    }
  }

#pragma unroll
  for (int nt = 0; nt < 8; ++nt)
#pragma unroll
    for (int r = 0; r < 8; ++r) {
      size_t row = (size_t)(rowBase + wave * 16 + r + hi * 8);
      out[row * DH + nt * 16 + m] = (_Float16)(acc[nt][r] * scale);
    }
}

// ---------------------------------------------------------------------------
// Kernel 2: flash attention.  One (batch, 128-query tile) per block.
// Each wave owns 16 query rows; loop over 64-key tiles, software-pipelined:
// tile t+1 global loads issue before tile t compute, commit to LDS after.
// ---------------------------------------------------------------------------
__global__ __launch_bounds__(256) void flash_attn_kernel(
    const _Float16* __restrict__ q, const _Float16* __restrict__ kmat,
    const _Float16* __restrict__ v, float* __restrict__ out) {
  __shared__ __align__(32) _Float16 Klds[64][128];     // K tile [key][d]
  __shared__ __align__(32) _Float16 Vt[128][64];       // V tile transposed [d][key]
  __shared__ __align__(32) _Float16 Plds[8][16][64];   // per-wave P staging [m][k]

  const int tid  = threadIdx.x;
  const int lane = tid & 31;
  const int wave = tid >> 5;
  const int m    = lane & 15;
  const int hi   = lane >> 4;
  const int qBase = blockIdx.y * SEQ + blockIdx.x * 128;  // flat row into [B*T, DH]
  const int bBase = blockIdx.y * SEQ;
  const int NT    = SEQ / 64;

  // ---- pipeline registers: one 64x128 f16 K tile + V tile, b128 granularity
  uint4 kreg[4], vreg[4];

  auto issue_loads = [&](int kt) {
    const int keyBase = bBase + kt * 64;
    const uint4* kg = (const uint4*)kmat;  // 16 uint4 per 128-half row
    const uint4* vg = (const uint4*)v;
#pragma unroll
    for (int j = 0; j < 4; ++j) {
      int i = tid + j * 256;               // 0..1023
      int key = i >> 4, q4 = i & 15;
      kreg[j] = kg[(size_t)(keyBase + key) * 16 + q4];
      vreg[j] = vg[(size_t)(keyBase + key) * 16 + q4];
    }
  };
  auto commit_tiles = [&]() {
#pragma unroll
    for (int j = 0; j < 4; ++j) {
      int i = tid + j * 256;
      int key = i >> 4, q4 = i & 15;       // q4 covers d = q4*8 .. q4*8+7
      ((uint4*)Klds)[key * 16 + q4] = kreg[j];
      union { uint4 u; _Float16 h[8]; } t; t.u = vreg[j];
#pragma unroll
      for (int e = 0; e < 8; ++e) Vt[q4 * 8 + e][key] = t.h[e];
    }
  };

  // Q fragments (16x128, as 4 A-frags of 16x32); 1/sqrt(DH) folded into Q.
  issue_loads(0);
  v16h qf[4];
  {
    const size_t qrow = (size_t)(qBase + wave * 16 + m);
#pragma unroll
    for (int kc = 0; kc < 4; ++kc) {
      v8u au;
#pragma unroll
      for (int p = 0; p < 8; ++p) {
        int k = kc * 32 + hi * 16 + p * 2;
        au[p] = *(const unsigned int*)&q[qrow * DH + k];
      }
      qf[kc] = __builtin_bit_cast(v16h, au);
    }
  }

  float mrow[8], lrow[8];
#pragma unroll
  for (int r = 0; r < 8; ++r) { mrow[r] = -1e30f; lrow[r] = 0.0f; }
  v8f o[8] = {};

  commit_tiles();
  __syncthreads();

  for (int kt = 0; kt < NT; ++kt) {
    if (kt + 1 < NT) issue_loads(kt + 1);  // overlap HBM latency with compute

    // S = Q * K^T : kc-outer so each A fragment is reused by 4 WMMAs
    v8f s[4] = {};
#pragma unroll
    for (int kc = 0; kc < 4; ++kc)
#pragma unroll
      for (int nt = 0; nt < 4; ++nt) {
        v16h b = *(const v16h*)&Klds[nt * 16 + m][kc * 32 + hi * 16];
        s[nt] = wmma_f16(qf[kc], b, s[nt]);
      }

    // Online softmax. Row of element r is M = r + 8*hi; its 16 key-values per
    // N-tile live in one VGPR across a 16-lane half -> shfl_xor 1/2/4/8 reduce.
#pragma unroll
    for (int r = 0; r < 8; ++r) {
      float tm = fmaxf(fmaxf(s[0][r], s[1][r]), fmaxf(s[2][r], s[3][r]));
#pragma unroll
      for (int d = 1; d < 16; d <<= 1) tm = fmaxf(tm, __shfl_xor(tm, d, 32));
      float mnew  = fmaxf(mrow[r], tm);
      float alpha = __expf(mrow[r] - mnew);
      mrow[r] = mnew;
      float ps = 0.0f;
#pragma unroll
      for (int nt = 0; nt < 4; ++nt) {
        float p = __expf(s[nt][r] - mnew);
        s[nt][r] = p;
        ps += p;
      }
#pragma unroll
      for (int d = 1; d < 16; d <<= 1) ps += __shfl_xor(ps, d, 32);
      lrow[r] = lrow[r] * alpha + ps;
#pragma unroll
      for (int ot = 0; ot < 8; ++ot) o[ot][r] *= alpha;
    }

    // Re-layout P: D-layout -> A-layout via wave-private LDS staging
#pragma unroll
    for (int nt = 0; nt < 4; ++nt)
#pragma unroll
      for (int r = 0; r < 8; ++r)
        Plds[wave][r + hi * 8][nt * 16 + m] = (_Float16)s[nt][r];
    asm volatile("" ::: "memory");  // same-wave LDS ops are in-order (DScnt)

    v16h pf[2];
#pragma unroll
    for (int kc2 = 0; kc2 < 2; ++kc2) {
      v8u au;
#pragma unroll
      for (int p = 0; p < 8; ++p) {
        int k = kc2 * 32 + hi * 16 + p * 2;
        au[p] = *(const unsigned int*)&Plds[wave][m][k];
      }
      pf[kc2] = __builtin_bit_cast(v16h, au);
    }

    // O += P * V : kc2-outer so each P fragment is reused by 8 WMMAs
#pragma unroll
    for (int kc2 = 0; kc2 < 2; ++kc2)
#pragma unroll
      for (int ot = 0; ot < 8; ++ot) {
        v16h b = *(const v16h*)&Vt[ot * 16 + m][kc2 * 32 + hi * 16];
        o[ot] = wmma_f16(pf[kc2], b, o[ot]);
      }

    __syncthreads();                       // all waves done reading tile kt
    if (kt + 1 < NT) commit_tiles();       // publish tile kt+1
    __syncthreads();
  }

  float rinv[8];
#pragma unroll
  for (int r = 0; r < 8; ++r) rinv[r] = 1.0f / lrow[r];
#pragma unroll
  for (int ot = 0; ot < 8; ++ot)
#pragma unroll
    for (int r = 0; r < 8; ++r) {
      size_t row = (size_t)(qBase + wave * 16 + r + hi * 8);
      out[row * DH + ot * 16 + m] = o[ot][r] * rinv[r];
    }
}

// ---------------------------------------------------------------------------
extern "C" void kernel_launch(void* const* d_in, const int* in_sizes, int n_in,
                              void* d_out, int out_size, void* d_ws, size_t ws_size,
                              hipStream_t stream) {
  (void)in_sizes; (void)n_in; (void)out_size; (void)ws_size;
  const float* x  = (const float*)d_in[0];
  const float* Wq = (const float*)d_in[1];
  const float* Wk = (const float*)d_in[2];
  const float* Wv = (const float*)d_in[3];

  _Float16* ws = (_Float16*)d_ws;
  const size_t N = (size_t)ROWS * DH;  // elements per Q/K/V buffer
  _Float16* qws = ws;
  _Float16* kws = ws + N;
  _Float16* vws = ws + 2 * N;

  const float scale = 0.08838834764831845f;  // 1/sqrt(DH), folded into Q

  dim3 blk(256);
  dim3 g1(ROWS / 128);
  qkv_proj_kernel<<<g1, blk, 0, stream>>>(x, Wq, qws, scale);
  qkv_proj_kernel<<<g1, blk, 0, stream>>>(x, Wk, kws, 1.0f);
  qkv_proj_kernel<<<g1, blk, 0, stream>>>(x, Wv, vws, 1.0f);

  dim3 g2(SEQ / 128, BATCH);
  flash_attn_kernel<<<g2, blk, 0, stream>>>(qws, kws, vws, (float*)d_out);
}
